// DEMFM_72705206386872
// MI455X (gfx1250) — matrix operand order repaired and checked by reading, not compiled, not run on hardware
//
#include <hip/hip_runtime.h>

typedef __attribute__((ext_vector_type(16))) _Float16 v16h;
typedef __attribute__((ext_vector_type(8)))  _Float16 v8h;
typedef __attribute__((ext_vector_type(8)))  float    v8f;
typedef __attribute__((ext_vector_type(4)))  float    v4f;
typedef __attribute__((ext_vector_type(4)))  int      v4i;

#define DEV __device__ __forceinline__

// Async global->LDS (CDNA5 GLOBAL_LOAD_ASYNC_TO_LDS_B128, ASYNCcnt-tracked),
// guarded so the file still compiles if this toolchain lacks the builtins.
#if defined(__has_builtin)
#if __has_builtin(__builtin_amdgcn_global_load_async_to_lds_b128) && \
    __has_builtin(__builtin_amdgcn_s_wait_asynccnt)
#define HAVE_ASYNC_LDS 1
#endif
#endif
#ifndef HAVE_ASYNC_LDS
#define HAVE_ASYNC_LDS 0
#endif

#if HAVE_ASYNC_LDS
// builtin signature (from hipcc diagnostic): (v4i AS1*, v4i AS3*, imm offset, imm cpol)
DEV void async_copy16(const float* g, float* l) {
    __builtin_amdgcn_global_load_async_to_lds_b128(
        (__attribute__((address_space(1))) v4i*)g,
        (__attribute__((address_space(3))) v4i*)l,
        0, 0);
}
#endif

DEV float siluf(float x)     { return x / (1.f + __expf(-x)); }
DEV float geluf(float x)     { float t = tanhf(0.7978845608028654f*(x + 0.044715f*x*x*x)); return 0.5f*x*(1.f + t); }
DEV float softplusf(float x) { return (x > 20.f) ? x : log1pf(__expf(x)); }

DEV float apply_act(float v, int act) {
    switch (act) {
        case 1: return siluf(v);
        case 2: return geluf(v);
        case 3: return softplusf(v);
        case 4: return fmaxf(v, 0.f);
        default: return v;
    }
}

// pack 8 consecutive f32 -> v8h
DEV v8h pack8(const float* s) {
    v4f f0 = *reinterpret_cast<const v4f*>(s);
    v4f f1 = *reinterpret_cast<const v4f*>(s + 4);
    v8h h;
    h[0] = (_Float16)f0[0]; h[1] = (_Float16)f0[1];
    h[2] = (_Float16)f0[2]; h[3] = (_Float16)f0[3];
    h[4] = (_Float16)f1[0]; h[5] = (_Float16)f1[1];
    h[6] = (_Float16)f1[2]; h[7] = (_Float16)f1[3];
    return h;
}

// ---------------------------------------------------------------------------
// Generic WMMA GEMM:  C[M,N(real)] = act( A[M,K]*W[N,K]^T * scale + bias ) (+C)
// A row-major (lda), W row-major out x in (ldw), C row-major (ldc), f32 I/O,
// v_wmma_f32_16x16x32_f16 compute. Block = 128 threads = 4 wave32 waves,
// 64x64 block tile. Fragment-layout LDS (ISA 7.12.2); aligned full tiles take
// a double-buffered GLOBAL_LOAD_ASYNC_TO_LDS_B128 staging pipeline.
//
// Fragment address math (8-wide k-octet "ko" is contiguous in a lane's frag):
//   A: lane = (m&15) + ((ko&8)?16:0), elems (ko&16)?8..15:0..7
//   B: lane = (n&15) + ((ko&8)?16:0), elems (ko&16)?8..15:0..7
// ---------------------------------------------------------------------------
__global__ __launch_bounds__(128)
void k_gemm(const float* __restrict__ A, int lda,
            const float* __restrict__ Wt, int ldw,
            float* __restrict__ C, int ldc,
            int M, int Kd, int Nreal,
            const float* __restrict__ bias,
            const float* __restrict__ scale, float scale_fac,
            int act, int addC)
{
    __shared__ __align__(32) _Float16 lA[2048];
    __shared__ __align__(32) _Float16 lB[2048];
#if HAVE_ASYNC_LDS
    __shared__ __align__(16) float stA[2][2048];   // raw f32 64x32 tiles
    __shared__ __align__(16) float stB[2][2048];
#endif

    const int tid  = threadIdx.x;
    const int wv   = tid >> 5;
    const int lane = tid & 31;
    const int m0   = blockIdx.x * 64;
    const int n0   = blockIdx.y * 64;

    v8f acc[4] = {};

#if HAVE_ASYNC_LDS
    const bool fast = (m0 + 64 <= M) && (n0 + 64 <= Nreal) &&
                      ((Kd & 31) == 0) && ((lda & 3) == 0) && ((ldw & 3) == 0);
    if (fast) {
        const int nkt = Kd >> 5;
        // prologue: issue async stage of tile 0 into buffer 0
        for (int e = tid; e < 512; e += 128) {
            int ml = e >> 3, kq = (e & 7) << 2;
            async_copy16(A + (size_t)(m0 + ml) * lda + kq, &stA[0][ml * 32 + kq]);
            async_copy16(Wt + (size_t)(n0 + ml) * ldw + kq, &stB[0][ml * 32 + kq]);
        }
        for (int kt = 0; kt < nkt; ++kt) {
            int buf = kt & 1;
            __builtin_amdgcn_s_wait_asynccnt(0);
            __syncthreads();
            if (kt + 1 < nkt) {   // overlap next tile's DMA with convert+WMMA
                int kb = (kt + 1) << 5;
                for (int e = tid; e < 512; e += 128) {
                    int ml = e >> 3, kq = (e & 7) << 2;
                    async_copy16(A + (size_t)(m0 + ml) * lda + kb + kq,
                                 &stA[buf ^ 1][ml * 32 + kq]);
                    async_copy16(Wt + (size_t)(n0 + ml) * ldw + kb + kq,
                                 &stB[buf ^ 1][ml * 32 + kq]);
                }
            }
            // convert staged f32 -> f16 fragment layout (b128 in, b128 out)
            for (int e = tid; e < 256; e += 128) {
                int ml = e >> 2, ko = (e & 3) << 3;
                v8h ha = pack8(&stA[buf][ml * 32 + ko]);
                *reinterpret_cast<v8h*>(
                    &lA[(ml >> 4) * 512 + ((ml & 15) + ((ko & 8) ? 16 : 0)) * 16 + ((ko & 16) ? 8 : 0)]) = ha;
                v8h hb = pack8(&stB[buf][ml * 32 + ko]);
                *reinterpret_cast<v8h*>(
                    &lB[(ml >> 4) * 512 + ((ml & 15) + ((ko & 8) ? 16 : 0)) * 16 + ((ko & 16) ? 8 : 0)]) = hb;
            }
            __syncthreads();
            v16h a = *reinterpret_cast<const v16h*>(&lA[wv * 512 + lane * 16]);
#pragma unroll
            for (int j = 0; j < 4; ++j) {
                v16h b = *reinterpret_cast<const v16h*>(&lB[j * 512 + lane * 16]);
                acc[j] = __builtin_amdgcn_wmma_f32_16x16x32_f16(
                    false, a, false, b, (short)0, acc[j], false, false);
            }
            __syncthreads();
        }
    } else
#endif
    {
        for (int kb = 0; kb < Kd; kb += 32) {
            // vectorized-when-possible direct-global staging, guarded edges
            for (int e = tid; e < 256; e += 128) {
                int ml = e >> 2, ko = (e & 3) << 3;
                int m = m0 + ml;
                v8h ha;
                if (m < M && kb + ko + 8 <= Kd && (lda & 3) == 0) {
                    ha = pack8(A + (size_t)m * lda + kb + ko);
                } else {
#pragma unroll
                    for (int j = 0; j < 8; ++j) {
                        int k = kb + ko + j;
                        ha[j] = (_Float16)((m < M && k < Kd) ? A[(size_t)m * lda + k] : 0.f);
                    }
                }
                *reinterpret_cast<v8h*>(
                    &lA[(ml >> 4) * 512 + ((ml & 15) + ((ko & 8) ? 16 : 0)) * 16 + ((ko & 16) ? 8 : 0)]) = ha;

                int n = n0 + ml;
                v8h hb;
                if (n < Nreal && kb + ko + 8 <= Kd && (ldw & 3) == 0) {
                    hb = pack8(Wt + (size_t)n * ldw + kb + ko);
                } else {
#pragma unroll
                    for (int j = 0; j < 8; ++j) {
                        int k = kb + ko + j;
                        hb[j] = (_Float16)((n < Nreal && k < Kd) ? Wt[(size_t)n * ldw + k] : 0.f);
                    }
                }
                *reinterpret_cast<v8h*>(
                    &lB[(ml >> 4) * 512 + ((ml & 15) + ((ko & 8) ? 16 : 0)) * 16 + ((ko & 16) ? 8 : 0)]) = hb;
            }
            __syncthreads();
            v16h a = *reinterpret_cast<const v16h*>(&lA[wv * 512 + lane * 16]);
#pragma unroll
            for (int j = 0; j < 4; ++j) {
                v16h b = *reinterpret_cast<const v16h*>(&lB[j * 512 + lane * 16]);
                acc[j] = __builtin_amdgcn_wmma_f32_16x16x32_f16(
                    false, a, false, b, (short)0, acc[j], false, false);
            }
            __syncthreads();
        }
    }

    // ---- epilogue: scale/bias/act, optional residual accumulate ------------
    const int mbase = m0 + wv * 16 + ((lane >> 4) << 3);
#pragma unroll
    for (int j = 0; j < 4; ++j) {
        int n = n0 + j * 16 + (lane & 15);
        if (n >= Nreal) continue;
        float sc = scale ? scale[n] * scale_fac : 1.f;
        float bs = bias ? bias[n] : 0.f;
#pragma unroll
        for (int r = 0; r < 8; ++r) {
            int m = mbase + r;
            if (m < M) {
                float v = acc[j][r] * sc + bs;
                v = apply_act(v, act);
                size_t idx = (size_t)m * ldc + n;
                if (addC) v += C[idx];
                C[idx] = v;
            }
        }
    }
}

// ---------------------------------------------------------------------------
// LayerNorm over last dim (token-major), two-pass per thread.
// ---------------------------------------------------------------------------
__global__ void k_ln(const float* __restrict__ x, int ldx,
                     const float* __restrict__ w, const float* __restrict__ b,
                     float* __restrict__ out, int ldo, int T, int Cn)
{
    int t = blockIdx.x * blockDim.x + threadIdx.x;
    if (t >= T) return;
    const float* xr = x + (size_t)t * ldx;
    float mu = 0.f;
    for (int c = 0; c < Cn; ++c) mu += xr[c];
    mu /= Cn;
    float var = 0.f;
    for (int c = 0; c < Cn; ++c) { float d = xr[c] - mu; var += d * d; }
    var /= Cn;
    float inv = rsqrtf(var + 1e-5f);
    float* orow = out + (size_t)t * ldo;
    for (int c = 0; c < Cn; ++c) orow[c] = (xr[c] - mu) * inv * w[c] + b[c];
}

// depthwise 3x3 (SAME, zero pad) + bias + silu; input token-major [b,h,w,*ldx]
__global__ void k_dwconv_silu(const float* __restrict__ xz, int ldx,
                              const float* __restrict__ w, const float* __restrict__ b,
                              float* __restrict__ out, int Bn, int Hh, int Ww, int D)
{
    size_t i = (size_t)blockIdx.x * blockDim.x + threadIdx.x;
    size_t total = (size_t)Bn * Hh * Ww * D;
    if (i >= total) return;
    int d = i % D; size_t t = i / D;
    int wq = t % Ww; size_t t2 = t / Ww;
    int h = t2 % Hh; int bb = (int)(t2 / Hh);
    float acc = 0.f;
    for (int ky = 0; ky < 3; ++ky) {
        int hy = h + ky - 1; if (hy < 0 || hy >= Hh) continue;
        for (int kx = 0; kx < 3; ++kx) {
            int wx = wq + kx - 1; if (wx < 0 || wx >= Ww) continue;
            acc += w[(d * 3 + ky) * 3 + kx] *
                   xz[(size_t)((bb * Hh + hy) * Ww + wx) * ldx + d];
        }
    }
    out[i] = siluf(acc + b[d]);
}

// scatter xin (token-major hw order) into scan order for direction kdir
__global__ void k_make_xs(const float* __restrict__ xin, float* __restrict__ xs,
                          int Bn, int Hh, int Ww, int D, int kdir)
{
    size_t i = (size_t)blockIdx.x * blockDim.x + threadIdx.x;
    int L = Hh * Ww;
    size_t total = (size_t)Bn * L * D;
    if (i >= total) return;
    int d = i % D; size_t t = i / D;
    int l = t % L; int bb = (int)(t / L);
    int h = l / Ww, w = l % Ww;
    int s;
    if      (kdir == 0) s = l;
    else if (kdir == 1) s = w * Hh + h;
    else if (kdir == 2) s = L - 1 - l;
    else                s = L - 1 - (w * Hh + h);
    xs[((size_t)bb * L + s) * D + d] = xin[i];
}

// selective scan for one direction; accumulates into canonical-hw token-major y
__global__ void k_scan(const float* __restrict__ xs,
                       const float* __restrict__ dbl, int lddbl,
                       const float* __restrict__ dt,
                       const float* __restrict__ Alog,
                       const float* __restrict__ Dsk,
                       float* __restrict__ ytok,
                       int Bn, int L, int D, int Hh, int Ww, int kdir)
{
    int i = blockIdx.x * blockDim.x + threadIdx.x;
    if (i >= Bn * D) return;
    int d = i % D, bb = i / D;
    float Aj[16], h[16];
#pragma unroll
    for (int j = 0; j < 16; ++j) { Aj[j] = -__expf(Alog[d * 16 + j]); h[j] = 0.f; }
    float Dv = Dsk[d];
    for (int l = 0; l < L; ++l) {
        size_t row = (size_t)bb * L + l;
        if (l + 8 < L) {   // stream next steps (global_prefetch_b8)
            __builtin_prefetch(&dbl[(row + 8) * lddbl], 0, 0);
            __builtin_prefetch(&xs[(row + 8) * D + d], 0, 0);
            __builtin_prefetch(&dt[(row + 8) * D + d], 0, 0);
        }
        float u   = xs[row * D + d];
        float dtv = dt[row * D + d];
        const float* br = dbl + row * lddbl;
        float y = 0.f;
#pragma unroll
        for (int j = 0; j < 16; ++j) {
            h[j] = __expf(dtv * Aj[j]) * h[j] + dtv * u * br[4 + j];
            y += h[j] * br[20 + j];
        }
        y += u * Dv;
        int jj = (kdir >= 2) ? (L - 1 - l) : l;
        int lc;
        if (kdir & 1) { int w2 = jj / Hh, h2 = jj % Hh; lc = h2 * Ww + w2; }
        else lc = jj;
        ytok[((size_t)bb * L + lc) * D + d] += y;
    }
}

// y = LN(y; out_norm) * silu(z), in place; z = xz[:, 128:]
__global__ void k_comb(float* __restrict__ y,
                       const float* __restrict__ xz, int ldxz,
                       const float* __restrict__ w, const float* __restrict__ b,
                       int T, int D)
{
    int t = blockIdx.x * blockDim.x + threadIdx.x;
    if (t >= T) return;
    float* yr = y + (size_t)t * D;
    float mu = 0.f;
    for (int c = 0; c < D; ++c) mu += yr[c];
    mu /= D;
    float var = 0.f;
    for (int c = 0; c < D; ++c) { float d2 = yr[c] - mu; var += d2 * d2; }
    var /= D;
    float inv = rsqrtf(var + 1e-5f);
    const float* zr = xz + (size_t)t * ldxz + 128;
    for (int c = 0; c < D; ++c)
        yr[c] = ((yr[c] - mu) * inv * w[c] + b[c]) * siluf(zr[c]);
}

// NCHW -> token-major column block
__global__ void k_nchw_to_tok(const float* __restrict__ src, float* __restrict__ dst,
                              int ldd, int coloff, int Bn, int Cn, int L)
{
    size_t i = (size_t)blockIdx.x * blockDim.x + threadIdx.x;
    size_t total = (size_t)Bn * Cn * L;
    if (i >= total) return;
    int l = i % L; size_t t = i / L;
    int c = t % Cn; int bb = (int)(t / Cn);
    dst[(size_t)(bb * (size_t)L + l) * ldd + coloff + c] = src[i];
}

// token-major column copy (concat builder)
__global__ void k_copy_cols(float* __restrict__ dst, int ldd, int coloff,
                            const float* __restrict__ src, int lds_, int Cn, int T)
{
    size_t i = (size_t)blockIdx.x * blockDim.x + threadIdx.x;
    if (i >= (size_t)T * Cn) return;
    int c = i % Cn; size_t t = i / Cn;
    dst[t * ldd + coloff + c] = src[t * lds_ + c];
}

__global__ void k_zero(float* p, size_t n)
{
    size_t i = (size_t)blockIdx.x * blockDim.x + threadIdx.x;
    if (i < n) p[i] = 0.f;
}
__global__ void k_copyn(float* __restrict__ d, const float* __restrict__ s, size_t n)
{
    size_t i = (size_t)blockIdx.x * blockDim.x + threadIdx.x;
    if (i < n) d[i] = s[i];
}
__global__ void k_addn(float* __restrict__ d, const float* __restrict__ s, size_t n)
{
    size_t i = (size_t)blockIdx.x * blockDim.x + threadIdx.x;
    if (i < n) d[i] += s[i];
}

// hor: token (b,h,w2) w2 in [0,128); value = interleave(x1,x2) along W
__global__ void k_pack_hor(const float* __restrict__ x1, const float* __restrict__ x2,
                           float* __restrict__ dst)
{
    size_t i = (size_t)blockIdx.x * blockDim.x + threadIdx.x;
    if (i >= (size_t)8 * 64 * 128 * 64) return;
    int c = i % 64; size_t t = i / 64;
    int w2 = t % 128; size_t t2 = t / 128;
    int h = t2 % 64; int bb = (int)(t2 / 64);
    const float* s = (w2 & 1) ? x2 : x1;
    dst[i] = s[((size_t)(bb * 64 + c) * 64 + h) * 64 + (w2 >> 1)];
}
// ver (already transposed): token (b,p,q), p = w index, q = interleaved 2h
__global__ void k_pack_ver(const float* __restrict__ x1, const float* __restrict__ x2,
                           float* __restrict__ dst)
{
    size_t i = (size_t)blockIdx.x * blockDim.x + threadIdx.x;
    if (i >= (size_t)8 * 64 * 128 * 64) return;
    int c = i % 64; size_t t = i / 64;
    int q = t % 128; size_t t2 = t / 128;
    int p = t2 % 64; int bb = (int)(t2 / 64);
    const float* s = (q & 1) ? x2 : x1;
    dst[i] = s[((size_t)(bb * 64 + c) * 64 + (q >> 1)) * 64 + p];
}
__global__ void k_pack_abs(const float* __restrict__ x1, const float* __restrict__ x2,
                           float* __restrict__ dst)
{
    size_t i = (size_t)blockIdx.x * blockDim.x + threadIdx.x;
    if (i >= (size_t)8 * 4096 * 64) return;
    int c = i % 64; size_t t = i / 64;
    int l = t % 4096; int bb = (int)(t / 4096);
    size_t si = ((size_t)(bb * 64 + c)) * 4096 + l;
    dst[i] = fabsf(x1[si] - x2[si]);
}
// split interleaved-W vss result (tokens b,h,w2) into two hw token-major maps
__global__ void k_unpack_hor(const float* __restrict__ vtok,
                             float* __restrict__ o1, float* __restrict__ o2)
{
    size_t i = (size_t)blockIdx.x * blockDim.x + threadIdx.x;
    if (i >= (size_t)8 * 4096 * 64) return;
    int c = i % 64; size_t t = i / 64;
    int w = t % 64; size_t t2 = t / 64;
    int h = t2 % 64; int bb = (int)(t2 / 64);
    size_t base = ((size_t)(bb * 64 + h) * 128 + 2 * w) * 64 + c;
    o1[i] = vtok[base];
    o2[i] = vtok[base + 64];
}
__global__ void k_unpack_ver(const float* __restrict__ vtok,
                             float* __restrict__ o1, float* __restrict__ o2)
{
    size_t i = (size_t)blockIdx.x * blockDim.x + threadIdx.x;
    if (i >= (size_t)8 * 4096 * 64) return;
    int c = i % 64; size_t t = i / 64;
    int w = t % 64; size_t t2 = t / 64;
    int h = t2 % 64; int bb = (int)(t2 / 64);
    size_t base = ((size_t)(bb * 64 + w) * 128 + 2 * h) * 64 + c;
    o1[i] = vtok[base];
    o2[i] = vtok[base + 64];
}

// 3x3 conv (SAME) from token-major input + BN-ReLU, write NCHW d_out
__global__ void k_conv3(const float* __restrict__ x, const float* __restrict__ wc,
                        const float* __restrict__ g, const float* __restrict__ bt,
                        float fac, float* __restrict__ out)
{
    size_t i = (size_t)blockIdx.x * blockDim.x + threadIdx.x;
    if (i >= (size_t)8 * 64 * 64 * 64) return;
    int w = i & 63; int h = (i >> 6) & 63; int o = (i >> 12) & 63; int bb = (int)(i >> 18);
    float acc = 0.f;
    for (int ky = 0; ky < 3; ++ky) {
        int hy = h + ky - 1; if (hy < 0 || hy >= 64) continue;
        for (int kx = 0; kx < 3; ++kx) {
            int wx = w + kx - 1; if (wx < 0 || wx >= 64) continue;
            const float* xr = x + ((size_t)(bb * 4096 + hy * 64 + wx)) * 64;
            const float* wr = wc + (size_t)o * 64 * 9 + ky * 3 + kx;
            for (int c = 0; c < 64; ++c) acc += wr[c * 9] * xr[c];
        }
    }
    out[i] = fmaxf(acc * g[o] * fac + bt[o], 0.f);
}

// ---------------------------------------------------------------------------
static inline int cdiv(long a, long b) { return (int)((a + b - 1) / b); }

extern "C" void kernel_launch(void* const* d_in, const int* in_sizes, int n_in,
                              void* d_out, int out_size, void* d_ws, size_t ws_size,
                              hipStream_t stream)
{
    (void)in_sizes; (void)n_in; (void)out_size; (void)ws_size;
    const float* x1         = (const float*)d_in[0];
    const float* x2         = (const float*)d_in[1];
    const float* conv_cat_w = (const float*)d_in[2];
    const float* conv_cat_b = (const float*)d_in[3];
    const float* conv_pre_w = (const float*)d_in[4];
    const float* conv_pre_b = (const float*)d_in[5];
    const float* ln1_w      = (const float*)d_in[6];
    const float* ln1_b      = (const float*)d_in[7];
    const float* in_proj_w  = (const float*)d_in[8];
    const float* in_proj_b  = (const float*)d_in[9];
    const float* dconv_w    = (const float*)d_in[10];
    const float* dconv_b    = (const float*)d_in[11];
    const float* x_proj_w   = (const float*)d_in[12];
    const float* dt_proj_w  = (const float*)d_in[13];
    const float* dt_proj_b  = (const float*)d_in[14];
    const float* A_log      = (const float*)d_in[15];
    const float* Dskip      = (const float*)d_in[16];
    const float* out_norm_w = (const float*)d_in[17];
    const float* out_norm_b = (const float*)d_in[18];
    const float* out_proj_w = (const float*)d_in[19];
    const float* ln2_w      = (const float*)d_in[20];
    const float* ln2_b      = (const float*)d_in[21];
    const float* fc1_w      = (const float*)d_in[22];
    const float* fc1_b      = (const float*)d_in[23];
    const float* fc2_w      = (const float*)d_in[24];
    const float* fc2_b      = (const float*)d_in[25];
    const float* en_w       = (const float*)d_in[26];
    const float* dr_w       = (const float*)d_in[27];
    const float* outc_w     = (const float*)d_in[28];
    const float* bn_w       = (const float*)d_in[29];
    const float* bn_b       = (const float*)d_in[30];

    const int Bn = 8, L0 = 4096;
    const int T0 = Bn * L0;          // 32768 tokens (64x64 blocks)
    const int T1 = 2 * T0;           // 65536 tokens (64x128 blocks)
    const size_t Tm = (size_t)T1;

    float* base = (float*)d_ws;
    size_t off = 0;
    auto AL = [&](size_t n) { float* p = base + off; off += n; return p; };

    // persistent (token-major [T0,64])
    float* xcat_t  = AL((size_t)T0 * 64);
    float* xh1_t   = AL((size_t)T0 * 64);
    float* xh2_t   = AL((size_t)T0 * 64);
    float* xver1_t = AL((size_t)T0 * 64);
    float* xver2_t = AL((size_t)T0 * 64);
    float* v1_t    = AL((size_t)T0 * 64);
    float* v2_t    = AL((size_t)T0 * 64);
    float* xo_t    = AL((size_t)T0 * 64);
    float* xsub_t  = AL((size_t)T0 * 64);
    // scratch (reused across vss blocks)
    float* s_in   = AL(Tm * 128);
    float* s_xtok = AL(Tm * 64);
    float* s_tmp  = AL(Tm * 256);
    float* s_xz   = AL(Tm * 256);
    float* s_xin  = AL(Tm * 128);
    float* s_xs   = AL(Tm * 128);
    float* s_dbl  = AL(Tm * 48);
    float* s_dt   = AL(Tm * 128);
    float* s_y    = AL(Tm * 128);

    const int TPB = 256;
    auto EW = [&](size_t n) { return dim3(cdiv((long)n, TPB)); };

    // ---- one VSS block; x in/out token-major in s_xtok ---------------------
    auto vss = [&](int i, int Hq, int Wq) {
        int Lq = Hq * Wq, T = Bn * Lq;
        size_t nd = (size_t)T * 128;
        k_ln<<<EW(T), TPB, 0, stream>>>(s_xtok, 64, ln1_w + i * 64, ln1_b + i * 64,
                                        s_tmp, 64, T, 64);
        k_gemm<<<dim3(cdiv(T, 64), 4), 128, 0, stream>>>(
            s_tmp, 64, in_proj_w + (size_t)i * 256 * 64, 64, s_xz, 256,
            T, 64, 256, in_proj_b + i * 256, nullptr, 1.f, 0, 0);
        k_dwconv_silu<<<EW(nd), TPB, 0, stream>>>(
            s_xz, 256, dconv_w + (size_t)i * 128 * 9, dconv_b + i * 128,
            s_xin, Bn, Hq, Wq, 128);
        k_zero<<<EW(nd), TPB, 0, stream>>>(s_y, nd);
        for (int k = 0; k < 4; ++k) {
            int ik = i * 4 + k;
            k_make_xs<<<EW(nd), TPB, 0, stream>>>(s_xin, s_xs, Bn, Hq, Wq, 128, k);
            k_gemm<<<dim3(cdiv(T, 64), 1), 128, 0, stream>>>(
                s_xs, 128, x_proj_w + (size_t)ik * 36 * 128, 128, s_dbl, 48,
                T, 128, 36, nullptr, nullptr, 1.f, 0, 0);
            k_gemm<<<dim3(cdiv(T, 64), 2), 128, 0, stream>>>(
                s_dbl, 48, dt_proj_w + (size_t)ik * 128 * 4, 4, s_dt, 128,
                T, 4, 128, dt_proj_b + ik * 128, nullptr, 1.f, 3, 0);
            k_scan<<<dim3(cdiv(Bn * 128, 128)), 128, 0, stream>>>(
                s_xs, s_dbl, 48, s_dt, A_log + (size_t)ik * 128 * 16,
                Dskip + ik * 128, s_y, Bn, Lq, 128, Hq, Wq, k);
        }
        k_comb<<<EW(T), TPB, 0, stream>>>(s_y, s_xz, 256,
                                          out_norm_w + i * 128, out_norm_b + i * 128, T, 128);
        k_gemm<<<dim3(cdiv(T, 64), 1), 128, 0, stream>>>(
            s_y, 128, out_proj_w + (size_t)i * 64 * 128, 128, s_xtok, 64,
            T, 128, 64, nullptr, nullptr, 1.f, 0, 1);          // residual add
        k_ln<<<EW(T), TPB, 0, stream>>>(s_xtok, 64, ln2_w + i * 64, ln2_b + i * 64,
                                        s_tmp, 64, T, 64);
        k_gemm<<<dim3(cdiv(T, 64), 4), 128, 0, stream>>>(
            s_tmp, 64, fc1_w + (size_t)i * 256 * 64, 64, s_xz, 256,
            T, 64, 256, fc1_b + i * 256, nullptr, 1.f, 2, 0);  // +bias, gelu
        k_gemm<<<dim3(cdiv(T, 64), 1), 128, 0, stream>>>(
            s_xz, 256, fc2_w + (size_t)i * 64 * 256, 256, s_xtok, 64,
            T, 256, 64, fc2_b + i * 64, nullptr, 1.f, 0, 1);   // residual add
    };

    // ---- block 0: concat(x1,x2) -> conv_cat -> vss0 ------------------------
    k_nchw_to_tok<<<EW((size_t)T0 * 64), TPB, 0, stream>>>(x1, s_in, 128, 0, Bn, 64, L0);
    k_nchw_to_tok<<<EW((size_t)T0 * 64), TPB, 0, stream>>>(x2, s_in, 128, 64, Bn, 64, L0);
    k_gemm<<<dim3(cdiv(T0, 64), 1), 128, 0, stream>>>(
        s_in, 128, conv_cat_w, 128, s_xtok, 64, T0, 128, 64,
        conv_cat_b, nullptr, 1.f, 0, 0);
    vss(0, 64, 64);
    k_copyn<<<EW((size_t)T0 * 64), TPB, 0, stream>>>(xcat_t, s_xtok, (size_t)T0 * 64);

    // ---- block 1: horizontal interleave -> conv_pre0 -> vss1 ---------------
    k_pack_hor<<<EW((size_t)T1 * 64), TPB, 0, stream>>>(x1, x2, s_in);
    k_gemm<<<dim3(cdiv(T1, 64), 1), 128, 0, stream>>>(
        s_in, 64, conv_pre_w, 64, s_xtok, 64, T1, 64, 64,
        conv_pre_b, nullptr, 1.f, 0, 0);
    vss(1, 64, 128);
    k_unpack_hor<<<EW((size_t)T0 * 64), TPB, 0, stream>>>(s_xtok, xh1_t, xh2_t);

    // ---- block 2: vertical interleave (transposed) -> conv_pre1 -> vss2 ----
    k_pack_ver<<<EW((size_t)T1 * 64), TPB, 0, stream>>>(x1, x2, s_in);
    k_gemm<<<dim3(cdiv(T1, 64), 1), 128, 0, stream>>>(
        s_in, 64, conv_pre_w + 64 * 64, 64, s_xtok, 64, T1, 64, 64,
        conv_pre_b + 64, nullptr, 1.f, 0, 0);
    vss(2, 64, 128);
    k_unpack_ver<<<EW((size_t)T0 * 64), TPB, 0, stream>>>(s_xtok, xver1_t, xver2_t);

    // ---- block 3: |x1-x2| -> conv_pre2 -> vss3 -----------------------------
    k_pack_abs<<<EW((size_t)T0 * 64), TPB, 0, stream>>>(x1, x2, s_in);
    k_gemm<<<dim3(cdiv(T0, 64), 1), 128, 0, stream>>>(
        s_in, 64, conv_pre_w + 2 * 64 * 64, 64, s_xtok, 64, T0, 64, 64,
        conv_pre_b + 2 * 64, nullptr, 1.f, 0, 0);
    vss(3, 64, 64);
    k_copyn<<<EW((size_t)T0 * 64), TPB, 0, stream>>>(xsub_t, s_xtok, (size_t)T0 * 64);

    // ---- encoder + decision with fused BN-ReLU -----------------------------
    const float fac = 0.99999500003749f;  // rsqrt(1 + 1e-5)
    // x_v1 = relu(bn0(en0 @ [xh1, xver1, x1]))
    k_copy_cols<<<EW((size_t)T0 * 64), TPB, 0, stream>>>(s_tmp, 192, 0, xh1_t, 64, 64, T0);
    k_copy_cols<<<EW((size_t)T0 * 64), TPB, 0, stream>>>(s_tmp, 192, 64, xver1_t, 64, 64, T0);
    k_nchw_to_tok<<<EW((size_t)T0 * 64), TPB, 0, stream>>>(x1, s_tmp, 192, 128, Bn, 64, L0);
    k_gemm<<<dim3(cdiv(T0, 64), 1), 128, 0, stream>>>(
        s_tmp, 192, en_w, 192, v1_t, 64, T0, 192, 64, bn_b, bn_w, fac, 4, 0);
    // x_v2 = relu(bn1(en1 @ [xh2, xver2, x2]))
    k_copy_cols<<<EW((size_t)T0 * 64), TPB, 0, stream>>>(s_tmp, 192, 0, xh2_t, 64, 64, T0);
    k_copy_cols<<<EW((size_t)T0 * 64), TPB, 0, stream>>>(s_tmp, 192, 64, xver2_t, 64, 64, T0);
    k_nchw_to_tok<<<EW((size_t)T0 * 64), TPB, 0, stream>>>(x2, s_tmp, 192, 128, Bn, 64, L0);
    k_gemm<<<dim3(cdiv(T0, 64), 1), 128, 0, stream>>>(
        s_tmp, 192, en_w + (size_t)64 * 192, 192, v2_t, 64, T0, 192, 64,
        bn_b + 64, bn_w + 64, fac, 4, 0);
    // xo = relu(bn2(dr @ [x_v1, x_v2, x_cat]))
    k_copy_cols<<<EW((size_t)T0 * 64), TPB, 0, stream>>>(s_tmp, 192, 0, v1_t, 64, 64, T0);
    k_copy_cols<<<EW((size_t)T0 * 64), TPB, 0, stream>>>(s_tmp, 192, 64, v2_t, 64, 64, T0);
    k_copy_cols<<<EW((size_t)T0 * 64), TPB, 0, stream>>>(s_tmp, 192, 128, xcat_t, 64, 64, T0);
    k_gemm<<<dim3(cdiv(T0, 64), 1), 128, 0, stream>>>(
        s_tmp, 192, dr_w, 192, xo_t, 64, T0, 192, 64,
        bn_b + 2 * 64, bn_w + 2 * 64, fac, 4, 0);

    // ---- final: 3x3 conv(xo + x_sub) + BN-ReLU -> d_out (NCHW) -------------
    k_addn<<<EW((size_t)T0 * 64), TPB, 0, stream>>>(xo_t, xsub_t, (size_t)T0 * 64);
    k_conv3<<<EW((size_t)8 * 64 * 64 * 64), TPB, 0, stream>>>(
        xo_t, outc_w, bn_w + 3 * 64, bn_b + 3 * 64, fac, (float*)d_out);
}
